// LayerNormMLP_3092376453258
// MI455X (gfx1250) — matrix operand order repaired and checked by reading, not compiled, not run on hardware
//
#include <hip/hip_runtime.h>
#include <hip/hip_bf16.h>
#include <math.h>

typedef __bf16 bf16_t;
typedef __attribute__((ext_vector_type(16))) __bf16 v16bf;
typedef __attribute__((ext_vector_type(8)))  __bf16 v8bf;
typedef __attribute__((ext_vector_type(4)))  __bf16 v4bf;
typedef __attribute__((ext_vector_type(8)))  float  v8f;

#define LN_THREADS 256
#define BM 128            // rows per block tile
#define STAGEK 64         // K staged per barrier (2 WMMA sub-steps)
#define APITCH 72         // bf16 per LDS A row (64 k + pad; 144B rows, 16B aligned)
#define BPITCH 72         // bf16 per LDS B^T row (64 k + pad)

// --------------------------------------------------------- CDNA5 async ops
// global_load_async_to_lds_b128: vdst = wave-relative LDS byte offset,
// vaddr = 64-bit global address. Tracked by ASYNCcnt (ISA 15.18.3 op 98).
__device__ __forceinline__ void async_copy_b128(unsigned lds_off, const void* gaddr) {
  asm volatile("global_load_async_to_lds_b128 %0, %1, off"
               :: "v"(lds_off), "v"(gaddr) : "memory");
}
__device__ __forceinline__ void wait_asynccnt0() {
  asm volatile("s_wait_asynccnt 0x0" ::: "memory");
}

// hardware tanh (TRANS32 on CDNA5); ISA requires 1 independent op after TRANS
__device__ __forceinline__ float fast_tanh(float x) {
#if __has_builtin(__builtin_amdgcn_tanhf)
  return __builtin_amdgcn_tanhf(x);
#else
  float r;
  asm volatile("v_tanh_f32 %0, %1\n\tv_nop" : "=v"(r) : "v"(x));
  return r;
#endif
}
__device__ __forceinline__ float gelu_tanh(float x) {
  return 0.5f * x * (1.0f + fast_tanh(0.7978845608028654f * (x + 0.044715f * x * x * x)));
}

// ---------------------------------------------------------------- LayerNorm
__global__ __launch_bounds__(LN_THREADS)
void ln_kernel(const float* __restrict__ x, const float* __restrict__ gamma,
               const float* __restrict__ beta, float* __restrict__ ln_out,
               bf16_t* __restrict__ ln_bf, int H) {
  const int row = blockIdx.x;
  const float4* x4 = (const float4*)(x + (size_t)row * H);
  float4*       y4 = (float4*)(ln_out + (size_t)row * H);
  bf16_t*       br = ln_bf + (size_t)row * H;
  const int tid = threadIdx.x;
  const int nvec = H >> 2;

  float s1 = 0.f, s2 = 0.f;
  for (int i = tid; i < nvec; i += LN_THREADS) {
    float4 v = x4[i];
    s1 += v.x + v.y + v.z + v.w;
    s2 += v.x * v.x + v.y * v.y + v.z * v.z + v.w * v.w;
  }
  for (int o = 16; o > 0; o >>= 1) {          // wave32 butterfly
    s1 += __shfl_xor(s1, o, 32);
    s2 += __shfl_xor(s2, o, 32);
  }
  __shared__ float ws1[LN_THREADS / 32], ws2[LN_THREADS / 32];
  const int wave = tid >> 5, lane = tid & 31;
  if (lane == 0) { ws1[wave] = s1; ws2[wave] = s2; }
  __syncthreads();
  float t1 = 0.f, t2 = 0.f;
  for (int w = 0; w < LN_THREADS / 32; ++w) { t1 += ws1[w]; t2 += ws2[w]; }
  const float invH = 1.0f / (float)H;
  const float mu   = t1 * invH;
  const float var  = t2 * invH - mu * mu;
  const float rstd = rsqrtf(var + 1e-6f);

  const float4* g4 = (const float4*)gamma;
  const float4* b4 = (const float4*)beta;
  for (int i = tid; i < nvec; i += LN_THREADS) {
    float4 v = x4[i], g = g4[i], b = b4[i], y;
    y.x = (v.x - mu) * rstd * g.x + b.x;
    y.y = (v.y - mu) * rstd * g.y + b.y;
    y.z = (v.z - mu) * rstd * g.z + b.z;
    y.w = (v.w - mu) * rstd * g.w + b.w;
    y4[i] = y;
    v4bf p; p[0] = (bf16_t)y.x; p[1] = (bf16_t)y.y; p[2] = (bf16_t)y.z; p[3] = (bf16_t)y.w;
    *(v4bf*)(br + 4 * i) = p;
  }
}

// ------------------------------------------------------------ frag builders
// A 16x32 bf16 fragment at K sub-step s: lanes 0-15 rows 0..15 K{0..7,16..23};
// lanes 16-31 same rows K{8..15,24..31}.
__device__ __forceinline__ v16bf frag_a(const bf16_t* sA, int row, int hi, int s) {
  union { v16bf v; v4bf q[4]; } u;
  const bf16_t* p = sA + row * APITCH + s * 32 + hi * 8;
  u.q[0] = *(const v4bf*)(p);
  u.q[1] = *(const v4bf*)(p + 4);
  u.q[2] = *(const v4bf*)(p + 16);
  u.q[3] = *(const v4bf*)(p + 20);
  return u.v;
}
// B 32x16 fragment from transposed tile sB[col][k]:
// lanes 0-15: N=lane, K{0..7,16..23}; lanes 16-31: N=lane-16, K{8..15,24..31}.
__device__ __forceinline__ v16bf frag_b(const bf16_t* sB, int col, int hi, int s) {
  union { v16bf v; v4bf q[4]; } u;
  const bf16_t* p = sB + col * BPITCH + s * 32 + hi * 8;
  u.q[0] = *(const v4bf*)(p);
  u.q[1] = *(const v4bf*)(p + 4);
  u.q[2] = *(const v4bf*)(p + 16);
  u.q[3] = *(const v4bf*)(p + 20);
  return u.v;
}

// --------------------------------------------------- unified pipelined GEMM
// GATED=true : A[N,K]bf16 x W[K,2*NCOL]fp32, both branches per block,
//              z = gelu(h0)*h1 -> Zout bf16 [N,NCOL]. Block = 128 x 64 m-cols.
// GATED=false: A[N,K]bf16 x W[K,NCOL]fp32 -> Fout fp32 [N,NCOL]. 128 x 128.
template <bool GATED>
__global__ __launch_bounds__(256)
void gemm_kernel(const bf16_t* __restrict__ A, const float* __restrict__ W,
                 bf16_t* __restrict__ Zout, float* __restrict__ Fout,
                 int K, int NCOL) {
  __shared__ bf16_t sA[2][BM * APITCH];     // [buf][row][k0..63]
  __shared__ bf16_t sB[2][128 * BPITCH];    // [buf][col][k0..63] transposed

  const int tid   = threadIdx.x;
  const int wave  = tid >> 5, lane = tid & 31;
  const int rlane = lane & 15, hi = lane >> 4;
  const int row0  = blockIdx.x * BM;
  const int n0    = blockIdx.y * (GATED ? 64 : 128);
  const long ldw  = GATED ? 2L * (long)NCOL : (long)NCOL;

  // staging thread mapping (fixed per thread)
  const int chA = tid & 7;                       // A k-chunk (8 bf16) 0..7
  const int rA  = tid >> 3;                      // A row 0..31 (+32p)
  const int c4  = lane * 4;                      // B col group 0..124
  const int rB  = tid >> 5;                      // B k-row base 0..7 (+8p)
  const long gcol = GATED ? ((c4 < 64) ? (long)(n0 + c4)
                                       : (long)NCOL + n0 + (c4 - 64))
                          : (long)(n0 + c4);
  const float*  Wc = W + gcol;
  const bf16_t* Ar = A + (size_t)(row0 + rA) * K + chA * 8;

  v8f acc[8];
#pragma unroll
  for (int t = 0; t < 8; ++t)
#pragma unroll
    for (int i = 0; i < 8; ++i) acc[t][i] = 0.0f;

  float4 wreg[8];

  // issue: 4 async A b128 copies -> sA[buf], 8 B float4 global loads -> wreg
  auto issue = [&](int kk, int buf) {
#pragma unroll
    for (int p = 0; p < 4; ++p)
      async_copy_b128(
          (unsigned)(size_t)&sA[buf][(rA + 32 * p) * APITCH + chA * 8],
          Ar + kk + (size_t)(32 * p) * K);
    const float* wp = Wc + (size_t)(kk + rB) * ldw;
#pragma unroll
    for (int p = 0; p < 8; ++p) {
      wreg[p] = *(const float4*)wp;
      wp += 8 * ldw;
    }
    if (kk + STAGEK < K)
      __builtin_prefetch(Wc + (size_t)(kk + STAGEK + rB) * ldw, 0, 1);
  };
  // commit: fp32 -> bf16, transpose-store into sB[buf]
  auto commit = [&](int buf) {
#pragma unroll
    for (int p = 0; p < 8; ++p) {
      bf16_t* d = &sB[buf][c4 * BPITCH + rB + 8 * p];
      d[0 * BPITCH] = (bf16_t)wreg[p].x;
      d[1 * BPITCH] = (bf16_t)wreg[p].y;
      d[2 * BPITCH] = (bf16_t)wreg[p].z;
      d[3 * BPITCH] = (bf16_t)wreg[p].w;
    }
  };

  issue(0, 0);
  commit(0);

  int cur = 0;
  for (int kk = 0; kk < K; kk += STAGEK) {
    wait_asynccnt0();           // this wave's async A copies are in LDS
    __syncthreads();            // all waves' copies + B commits visible
    const bool hasNext = (kk + STAGEK) < K;
    if (hasNext) issue(kk + STAGEK, cur ^ 1);  // overlap with compute below

    const bf16_t* sAc = &sA[cur][0];
    const bf16_t* sBc = &sB[cur][0];
#pragma unroll
    for (int s = 0; s < 2; ++s) {
      const v16bf a = frag_a(sAc, wave * 16 + rlane, hi, s);
      // preload ALL 8 B fragments into distinct registers so the scheduler can
      // batch the ds_loads and emit partial dscnt waits before each WMMA
      v16bf b[8];
#pragma unroll
      for (int t = 0; t < 8; ++t)
        b[t] = frag_b(sBc, t * 16 + rlane, hi, s);
#pragma unroll
      for (int t = 0; t < 8; ++t)
        acc[t] = __builtin_amdgcn_wmma_f32_16x16x32_bf16(
            false, a, false, b[t], (short)0, acc[t], false, false);
    }

    if (hasNext) commit(cur ^ 1);              // loadcnt wait lands post-WMMA
    cur ^= 1;
  }

  // epilogue — C layout: VGPR i, lanes<16 -> row i, lanes>=16 -> row 8+i; col = lane&15
  const int or0 = row0 + wave * 16 + hi * 8;
  if (GATED) {
#pragma unroll
    for (int t = 0; t < 4; ++t) {
      const int zc = n0 + t * 16 + rlane;
#pragma unroll
      for (int i = 0; i < 8; ++i) {
        const float h0 = acc[t][i];
        const float h1 = acc[t + 4][i];
        Zout[(size_t)(or0 + i) * NCOL + zc] = (bf16_t)(gelu_tanh(h0) * h1);
      }
    }
  } else {
#pragma unroll
    for (int t = 0; t < 8; ++t) {
      const int oc = n0 + t * 16 + rlane;
#pragma unroll
      for (int i = 0; i < 8; ++i)
        Fout[(size_t)(or0 + i) * NCOL + oc] = acc[t][i];
    }
  }
}

// ------------------------------------------------------------------ launch
extern "C" void kernel_launch(void* const* d_in, const int* in_sizes, int n_in,
                              void* d_out, int out_size, void* d_ws, size_t ws_size,
                              hipStream_t stream) {
  const float* x     = (const float*)d_in[0];   // [S,B,H]
  const float* gamma = (const float*)d_in[1];   // [H]
  const float* beta  = (const float*)d_in[2];   // [H]
  const float* w1    = (const float*)d_in[3];   // [H,2,M]
  const float* w2    = (const float*)d_in[4];   // [M,H]

  const int H = in_sizes[1];
  const int N = in_sizes[0] / H;                // S*B tokens
  const int M = in_sizes[3] / (2 * H);

  float* out    = (float*)d_out;                // [N,H]
  float* ln_out = out + (size_t)N * H;          // [N,H] (second tuple output)

  bf16_t* ln_bf = (bf16_t*)d_ws;                // [N,H] bf16
  bf16_t* z_bf  = ln_bf + (size_t)N * H;        // [N,M] bf16

  ln_kernel<<<N, LN_THREADS, 0, stream>>>(x, gamma, beta, ln_out, ln_bf, H);

  dim3 g1(N / BM, M / 64);   // row tiles fastest -> L2 reuse of W1 strips
  gemm_kernel<true><<<g1, 256, 0, stream>>>(ln_bf, w1, z_bf, nullptr, H, M);

  dim3 g2(N / BM, H / 128);
  gemm_kernel<false><<<g2, 256, 0, stream>>>(z_bf, w2, nullptr, out, M, H);
}
